// MeshGraphNet_6055903888191
// MI455X (gfx1250) — matrix operand order repaired and checked by reading, not compiled, not run on hardware
//
#include <hip/hip_runtime.h>
#include <hip/hip_bf16.h>

// ---------------------------------------------------------------------------
// MeshGraphNet forward for MI455X (gfx1250): fused WMMA MLP kernels.
// ---------------------------------------------------------------------------

typedef _Float16 v16h __attribute__((ext_vector_type(16)));
typedef _Float16 v8h  __attribute__((ext_vector_type(8)));
typedef _Float16 v4h  __attribute__((ext_vector_type(4)));
typedef float    v8f  __attribute__((ext_vector_type(8)));

#define NNODES 20000
#define NEDGES 320000

__device__ inline v8f vzero8() {
  v8f z;
#pragma unroll
  for (int i = 0; i < 8; ++i) z[i] = 0.f;
  return z;
}

__device__ inline float lrelu(float x) { return x > 0.f ? x : 0.01f * x; }

union HV16 { v8h h[2]; v16h v; };

// Accumulate NT 16x16 output tiles: C[rt, ctBase..ctBase+NT-1] += A(rt,:) * B.
// lds_in: row-major f16 activations with row stride STRIDE, KTILES K-tiles.
// Wp: packed f16 weights, per-(ct,kt) tile of 512 f16 laid out so each lane's
//     16 values are contiguous (B-fragment layout: lanes 0-15 -> K 0..15,
//     lanes 16-31 -> K 16..31, N = lane%16).
template<int KTILES, int STRIDE, int NT>
__device__ inline void gemm_acc(const _Float16* lds_in,
                                const _Float16* __restrict__ Wp,
                                int rt, int ctBase, int lane, v8f* acc) {
  const int m   = rt * 16 + (lane & 15);
  const int khi = (lane >> 4) << 4;   // hi half-wave holds K 16..31
#pragma unroll
  for (int kt = 0; kt < KTILES; ++kt) {
    HV16 a;
    const _Float16* ap = lds_in + m * STRIDE + kt * 32 + khi;
    a.h[0] = *(const v8h*)(ap);        // K 0..7   (or 16..23)
    a.h[1] = *(const v8h*)(ap + 8);    // K 8..15  (or 24..31)
    if (kt + 1 < KTILES)               // -> global_prefetch_b8
      __builtin_prefetch(Wp + ((size_t)ctBase * KTILES + kt + 1) * 512 + lane * 16, 0, 3);
#pragma unroll
    for (int n = 0; n < NT; ++n) {
      const _Float16* bp = Wp + ((size_t)(ctBase + n) * KTILES + kt) * 512 + lane * 16;
      v16h b = *(const v16h*)bp;
      acc[n] = __builtin_amdgcn_wmma_f32_16x16x32_f16(
          false, a.v, false, b, (short)0, acc[n], false, false);
    }
  }
}

// Fused 3-layer MLP: X[32 x IN] -> lrelu(XW1+b1) -> lrelu(..W2+b2) -> ..W3+b3
// -> optional LayerNorm(gamma,beta) -> optional +resid -> out32 (+out16).
// MODE 0: x0 = dense f32 [nrows x in_f]   (encoders, zero-padded to 32k)
// MODE 1: x0 = e_f16, x1 = h_f16, concat [e | h[src] | h[dst]]   (edge MLP)
// MODE 2: x0 = h_f16, x1 = pe_f32, concat [h | pe]               (node MLP)
// MODE 3: x0 = dense f16 [nrows x 128]                           (decoder)
template<int IN_TILES, int OUT_TILES, bool LN, int MODE>
__global__ __launch_bounds__(256)
void mlp_fused(const void* __restrict__ x0, const void* __restrict__ x1,
               const int* __restrict__ srcIdx, const int* __restrict__ dstIdx,
               const _Float16* __restrict__ W1, const float* __restrict__ b1,
               const _Float16* __restrict__ W2, const float* __restrict__ b2,
               const _Float16* __restrict__ W3, const float* __restrict__ b3,
               const float* __restrict__ gamma, const float* __restrict__ beta,
               const float* __restrict__ resid,
               float* __restrict__ out32, _Float16* __restrict__ out16,
               int nrows, int in_f, int out_cols) {
  constexpr int LB = IN_TILES * 32;   // padded input width
  constexpr int OF = OUT_TILES * 16;  // padded output width
  __shared__ __align__(16) _Float16 xbuf[32 * LB];
  __shared__ __align__(16) _Float16 f1buf[32 * 256];
  __shared__ __align__(16) _Float16 f2buf[32 * 256];
  float* obuf = reinterpret_cast<float*>(f1buf);  // 32*OF floats <= 16KB

  const int tid  = threadIdx.x;
  const int lane = tid & 31;
  const int w    = tid >> 5;           // wave id 0..7
  const int row0 = blockIdx.x * 32;

  // ---- gather/convert input slab into LDS (f16) ----
  if (MODE == 0) {
    const float* X = (const float*)x0;
    for (int idx = tid; idx < 32 * LB; idx += 256) {
      int r = idx / LB, c = idx - r * LB;
      int g = row0 + r;
      float v = (g < nrows && c < in_f) ? X[(size_t)g * in_f + c] : 0.f;
      xbuf[idx] = (_Float16)v;
    }
  } else {
    constexpr int LB4 = LB / 4;
    for (int idx = tid; idx < 32 * LB4; idx += 256) {
      int r = idx / LB4, c = (idx - r * LB4) * 4;
      int g = row0 + r;
      v4h val;
      val.x = val.y = val.z = val.w = (_Float16)0.f;
      if (g < nrows) {
        if (MODE == 1) {
          const _Float16* e16 = (const _Float16*)x0;
          const _Float16* h16 = (const _Float16*)x1;
          if (c < 128)      val = *(const v4h*)&e16[(size_t)g * 128 + c];
          else if (c < 256) val = *(const v4h*)&h16[(size_t)srcIdx[g] * 128 + (c - 128)];
          else              val = *(const v4h*)&h16[(size_t)dstIdx[g] * 128 + (c - 256)];
        } else if (MODE == 2) {
          const _Float16* h16 = (const _Float16*)x0;
          const float*    pe  = (const float*)x1;
          if (c < 128) {
            val = *(const v4h*)&h16[(size_t)g * 128 + c];
          } else {
            const float* p = &pe[(size_t)g * 128 + (c - 128)];
            val.x = (_Float16)p[0]; val.y = (_Float16)p[1];
            val.z = (_Float16)p[2]; val.w = (_Float16)p[3];
          }
        } else {
          const _Float16* h16 = (const _Float16*)x0;
          val = *(const v4h*)&h16[(size_t)g * 128 + c];
        }
      }
      *(v4h*)&xbuf[r * LB + c] = val;
    }
  }
  __syncthreads();

  // ---- layer 1: [32 x LB] @ W1 -> f1 [32 x 256], bias + leaky ReLU ----
  {
    const int rt  = w & 1;
    const int ctB = (w >> 1) * 4;
    v8f acc[4];
#pragma unroll
    for (int n = 0; n < 4; ++n) acc[n] = vzero8();
    gemm_acc<IN_TILES, LB, 4>(xbuf, W1, rt, ctB, lane, acc);
    const int mB = rt * 16 + ((lane >> 4) << 3);
#pragma unroll
    for (int n = 0; n < 4; ++n) {
      const int col = (ctB + n) * 16 + (lane & 15);
      const float bv = b1[col];
#pragma unroll
      for (int j = 0; j < 8; ++j)
        f1buf[(mB + j) * 256 + col] = (_Float16)lrelu(acc[n][j] + bv);
    }
  }
  __syncthreads();

  // ---- layer 2: f1 @ W2 -> f2 [32 x 256], bias + leaky ReLU ----
  {
    const int rt  = w & 1;
    const int ctB = (w >> 1) * 4;
    v8f acc[4];
#pragma unroll
    for (int n = 0; n < 4; ++n) acc[n] = vzero8();
    gemm_acc<8, 256, 4>(f1buf, W2, rt, ctB, lane, acc);
    const int mB = rt * 16 + ((lane >> 4) << 3);
#pragma unroll
    for (int n = 0; n < 4; ++n) {
      const int col = (ctB + n) * 16 + (lane & 15);
      const float bv = b2[col];
#pragma unroll
      for (int j = 0; j < 8; ++j)
        f2buf[(mB + j) * 256 + col] = (_Float16)lrelu(acc[n][j] + bv);
    }
  }
  __syncthreads();

  // ---- layer 3: f2 @ W3 -> obuf [32 x OF] f32 (aliases f1 LDS) ----
  {
    constexpr int T   = 2 * OUT_TILES;
    constexpr int TPW = (T + 7) / 8;
#pragma unroll
    for (int t = 0; t < TPW; ++t) {
      const int  id     = w * TPW + t;
      const bool active = id < T;
      const int  idm    = active ? id : 0;
      const int  rt = idm & 1, ct = idm >> 1;
      v8f acc[1];
      acc[0] = vzero8();
      gemm_acc<8, 256, 1>(f2buf, W3, rt, ct, lane, acc);
      if (active) {
        const int col = ct * 16 + (lane & 15);
        const float bv = (col < out_cols) ? b3[col] : 0.f;
        const int mB = rt * 16 + ((lane >> 4) << 3);
#pragma unroll
        for (int j = 0; j < 8; ++j)
          obuf[(mB + j) * OF + col] = acc[0][j] + bv;
      }
    }
  }
  __syncthreads();

  // ---- LayerNorm + residual + store (one row per thread) ----
  for (int r = tid; r < 32; r += 256) {
    const int g = row0 + r;
    if (g >= nrows) continue;
    float* rp = obuf + r * OF;
    if (LN) {
      float mu = 0.f;
      for (int c = 0; c < OF; ++c) mu += rp[c];
      mu *= (1.f / OF);
      float var = 0.f;
      for (int c = 0; c < OF; ++c) { float d = rp[c] - mu; var += d * d; }
      var *= (1.f / OF);
      const float rs = rsqrtf(var + 1e-5f);
      for (int c = 0; c < OF; ++c) {
        float vv = (rp[c] - mu) * rs * gamma[c] + beta[c];
        if (resid) vv += resid[(size_t)g * OF + c];
        out32[(size_t)g * OF + c] = vv;
        out16[(size_t)g * OF + c] = (_Float16)vv;
      }
    } else {
      for (int c = 0; c < out_cols; ++c)
        out32[(size_t)g * out_cols + c] = rp[c];
    }
  }
}

// Pack fp32 weight [K x N] (row-major) into f16 WMMA B-fragment tiles:
// tile (ct, kt) holds 512 f16; lane l's 16 values contiguous:
// k = kt*32 + (l>=16 ? 16 : 0) + j, n = ct*16 + (l & 15). Zero-pad OOB.
__global__ void pack_w(const float* __restrict__ W, _Float16* __restrict__ out,
                       int K, int N, int Ktiles, int Ntiles) {
  const int total = Ktiles * Ntiles * 512;
  for (int i = blockIdx.x * blockDim.x + threadIdx.x; i < total;
       i += gridDim.x * blockDim.x) {
    const int r = i & 511, t = i >> 9;
    const int lane = r >> 4, j = r & 15;
    const int ct = t / Ktiles, kt = t - ct * Ktiles;
    const int k = kt * 32 + ((lane >> 4) << 4) + j;
    const int n = ct * 16 + (lane & 15);
    out[i] = (k < K && n < N) ? (_Float16)W[(size_t)k * N + n] : (_Float16)0.f;
  }
}

__global__ void fill_zero4(float* __restrict__ p, int n4) {
  int i = blockIdx.x * blockDim.x + threadIdx.x;
  if (i < n4) {
    float4 z; z.x = z.y = z.z = z.w = 0.f;
    ((float4*)p)[i] = z;
  }
}

// segment_sum: pe[dst[e], :] += e32[e, :]  (32 threads/edge, float4 each)
__global__ void scatter_add(const float* __restrict__ e32,
                            const int* __restrict__ dstIdx,
                            float* __restrict__ pe, int E) {
  const int i = blockIdx.x * blockDim.x + threadIdx.x;
  const int edge = i >> 5;
  const int c = (i & 31) * 4;
  if (edge < E) {
    const float4 v = *(const float4*)&e32[(size_t)edge * 128 + c];
    float* base = &pe[(size_t)dstIdx[edge] * 128 + c];
    unsafeAtomicAdd(base + 0, v.x);
    unsafeAtomicAdd(base + 1, v.y);
    unsafeAtomicAdd(base + 2, v.z);
    unsafeAtomicAdd(base + 3, v.w);
  }
}

// ---------------------------------------------------------------------------
// Host side
// ---------------------------------------------------------------------------

struct MlpP { const float *Wi, *bi, *Wh, *bh, *Wo, *bo, *gamma, *beta; };

// Resolve one MLP param group from d_in using in_sizes; supports both
// insertion-order {Wi,bi,Wh,bh,Wo,bo,gamma,beta} and pytree-alphabetical
// {Wh,Wi,Wo,beta,bh,bi,bo,gamma} flattening.
static void resolve(void* const* din, const int* sz, int base,
                    int in_f, int out_f, bool ln, MlpP& p) {
  const int n = ln ? 8 : 6;
  long expIns[8] = { (long)in_f * 256, 256, 65536, 256,
                     (long)256 * out_f, out_f, out_f, out_f };
  bool ins = true;
  for (int i = 0; i < n; ++i)
    if ((long)sz[base + i] != expIns[i]) { ins = false; break; }
  const float* a[8];
  for (int i = 0; i < 8; ++i) a[i] = (const float*)din[base + (i < n ? i : 0)];
  if (ins) {
    p.Wi = a[0]; p.bi = a[1]; p.Wh = a[2]; p.bh = a[3];
    p.Wo = a[4]; p.bo = a[5];
    p.gamma = ln ? a[6] : nullptr; p.beta = ln ? a[7] : nullptr;
  } else if (ln) {  // alpha: Wh,Wi,Wo,beta,bh,bi,bo,gamma
    p.Wh = a[0]; p.Wi = a[1]; p.Wo = a[2]; p.beta = a[3];
    p.bh = a[4]; p.bi = a[5]; p.bo = a[6]; p.gamma = a[7];
  } else {          // alpha: Wh,Wi,Wo,bh,bi,bo
    p.Wh = a[0]; p.Wi = a[1]; p.Wo = a[2];
    p.bh = a[3]; p.bi = a[4]; p.bo = a[5];
    p.gamma = nullptr; p.beta = nullptr;
  }
}

struct PackedMlp { _Float16 *W1, *W2, *W3; int kt1, nt3; };

extern "C" void kernel_launch(void* const* d_in, const int* in_sizes, int n_in,
                              void* d_out, int out_size, void* d_ws, size_t ws_size,
                              hipStream_t stream) {
  (void)n_in; (void)out_size; (void)ws_size;
  const int N = NNODES, E = NEDGES;
  const float* nfeat = (const float*)d_in[0];
  const float* efeat = (const float*)d_in[1];
  const int*   src   = (const int*)d_in[2];
  const int*   dst   = (const int*)d_in[3];

  MlpP encN, encE, pEdge[3], pNode[3], outP;
  resolve(d_in, in_sizes, 4, 12, 128, true, encN);
  resolve(d_in, in_sizes, 12, 4, 128, true, encE);
  for (int i = 0; i < 3; ++i) resolve(d_in, in_sizes, 20 + 8 * i, 384, 128, true, pEdge[i]);
  for (int i = 0; i < 3; ++i) resolve(d_in, in_sizes, 44 + 8 * i, 256, 128, true, pNode[i]);
  resolve(d_in, in_sizes, 68, 128, 2, false, outP);

  // workspace bump allocator (256B aligned)
  char* base = (char*)d_ws;
  size_t off = 0;
  auto bump = [&](size_t bytes) -> void* {
    off = (off + 255) & ~(size_t)255;
    void* p = base + off;
    off += bytes;
    return p;
  };

  float*    h32 = (float*)bump((size_t)N * 128 * 4);
  _Float16* h16 = (_Float16*)bump((size_t)N * 128 * 2);
  float*    e32 = (float*)bump((size_t)E * 128 * 4);
  _Float16* e16 = (_Float16*)bump((size_t)E * 128 * 2);
  float*    pe  = (float*)bump((size_t)N * 128 * 4);

  auto packMlp = [&](const MlpP& p, int in_f, int out_f) -> PackedMlp {
    PackedMlp q;
    q.kt1 = (in_f + 31) / 32;
    q.nt3 = (out_f + 15) / 16;
    q.W1 = (_Float16*)bump((size_t)q.kt1 * 16 * 512 * 2);
    q.W2 = (_Float16*)bump((size_t)8 * 16 * 512 * 2);
    q.W3 = (_Float16*)bump((size_t)8 * q.nt3 * 512 * 2);
    int t1 = q.kt1 * 16 * 512;
    int t2 = 8 * 16 * 512;
    int t3 = 8 * q.nt3 * 512;
    pack_w<<<(t1 + 255) / 256, 256, 0, stream>>>(p.Wi, q.W1, in_f, 256, q.kt1, 16);
    pack_w<<<(t2 + 255) / 256, 256, 0, stream>>>(p.Wh, q.W2, 256, 256, 8, 16);
    pack_w<<<(t3 + 255) / 256, 256, 0, stream>>>(p.Wo, q.W3, 256, out_f, 8, q.nt3);
    return q;
  };

  PackedMlp qEncN = packMlp(encN, 12, 128);
  PackedMlp qEncE = packMlp(encE, 4, 128);
  PackedMlp qEdge[3], qNode[3];
  for (int i = 0; i < 3; ++i) qEdge[i] = packMlp(pEdge[i], 384, 128);
  for (int i = 0; i < 3; ++i) qNode[i] = packMlp(pNode[i], 256, 128);
  PackedMlp qOut = packMlp(outP, 128, 2);

  const int NB = (N + 31) / 32;   // 625
  const int EB = (E + 31) / 32;   // 10000

  // encoders: h = MLP(nfeat), e = MLP(efeat)
  mlp_fused<1, 8, true, 0><<<NB, 256, 0, stream>>>(
      nfeat, nullptr, nullptr, nullptr,
      qEncN.W1, encN.bi, qEncN.W2, encN.bh, qEncN.W3, encN.bo,
      encN.gamma, encN.beta, nullptr, h32, h16, N, 12, 128);
  mlp_fused<1, 8, true, 0><<<EB, 256, 0, stream>>>(
      efeat, nullptr, nullptr, nullptr,
      qEncE.W1, encE.bi, qEncE.W2, encE.bh, qEncE.W3, encE.bo,
      encE.gamma, encE.beta, nullptr, e32, e16, E, 4, 128);

  // message-passing iterations
  for (int it = 0; it < 3; ++it) {
    // e = MLP([e | h[src] | h[dst]]) + e
    mlp_fused<12, 8, true, 1><<<EB, 256, 0, stream>>>(
        e16, h16, src, dst,
        qEdge[it].W1, pEdge[it].bi, qEdge[it].W2, pEdge[it].bh,
        qEdge[it].W3, pEdge[it].bo, pEdge[it].gamma, pEdge[it].beta,
        e32, e32, e16, E, 384, 128);
    // pe = segment_sum(e, dst)
    int n4 = N * 128 / 4;
    fill_zero4<<<(n4 + 255) / 256, 256, 0, stream>>>(pe, n4);
    int sthreads = E * 32;
    scatter_add<<<(sthreads + 255) / 256, 256, 0, stream>>>(e32, dst, pe, E);
    // h = MLP([h | pe]) + h
    mlp_fused<8, 8, true, 2><<<NB, 256, 0, stream>>>(
        h16, pe, nullptr, nullptr,
        qNode[it].W1, pNode[it].bi, qNode[it].W2, pNode[it].bh,
        qNode[it].W3, pNode[it].bo, pNode[it].gamma, pNode[it].beta,
        h32, h32, h16, N, 256, 128);
  }

  // decoder: out = MLP(h)  [N x 2], no LN
  mlp_fused<4, 1, false, 3><<<NB, 256, 0, stream>>>(
      h16, nullptr, nullptr, nullptr,
      qOut.W1, outP.bi, qOut.W2, outP.bh, qOut.W3, outP.bo,
      nullptr, nullptr, nullptr, (float*)d_out, nullptr, N, 128, 2);
}
